// RITS_17712445129440
// MI455X (gfx1250) — matrix operand order repaired and checked by reading, not compiled, not run on hardware
//
#include <hip/hip_runtime.h>
#include <hip/hip_bf16.h>

// Problem constants (reference: B,T,D,H = 512,128,128,256)
#define B_ 512
#define T_ 128
#define D_ 128
#define H_ 256

typedef __attribute__((ext_vector_type(16))) __bf16 v16bf;
typedef __attribute__((ext_vector_type(8)))  float  v8f;

__device__ __forceinline__ v8f wmma_bf16(v16bf a, v16bf b, v8f c) {
  // D = A(16x32) * B(32x16) + C, f32 accumulate
  return __builtin_amdgcn_wmma_f32_16x16x32_bf16(
      /*neg_a=*/false, a, /*neg_b=*/false, b,
      /*c_mod=*/(short)0, c, /*reuse_a=*/false, /*reuse_b=*/false);
}

union Pack32 { uint4 q[2]; v16bf v; };

// A-operand (16xK row-major bf16 panel in LDS). ISA 16-bit A layout:
// lane<16 : K[k0+0..8)  and K[k0+16..24)
// lane>=16: K[k0+8..16) and K[k0+24..32)
__device__ __forceinline__ v16bf load_a(const __bf16* row, int hi, int k0) {
  Pack32 u;
  u.q[0] = *(const uint4*)(row + k0 + hi * 8);
  u.q[1] = *(const uint4*)(row + k0 + 16 + hi * 8);
  return u.v;
}

// B-operand from global weight stored row-major W[N][K] (== B column-major).
// lane<16 : K[k0+0..8) and K[k0+16..24), lane>=16: K[k0+8..16) and K[k0+24..32) of column n.
__device__ __forceinline__ v16bf load_b(const __bf16* colp, int hi, int k0) {
  Pack32 u;
  u.q[0] = *(const uint4*)(colp + k0 + hi * 8);
  u.q[1] = *(const uint4*)(colp + k0 + 16 + hi * 8);
  return u.v;
}

__device__ __forceinline__ float sigm(float x) {
  return 1.f / (1.f + __expf(-x));
}

// Anti-LICM that preserves address-space inference: the weight POINTERS stay
// as kernargs (=> global_load saddr form), but a 32-bit offset becomes opaque
// per phase so the loads are formally loop-variant and cannot be hoisted out
// of the T loop (which previously caused scratch spilling).
#define OPQ(x) asm volatile("" : "+s"(x))

// ---------------- prep: fp32 weights -> bf16 (+ masks, fused biases) --------
__global__ void rits_prep_kernel(const float* __restrict__ Wtdh,
                                 const float* __restrict__ Whist,
                                 const float* __restrict__ Wfeat,
                                 const float* __restrict__ Wcomb,
                                 const float* __restrict__ Wih,
                                 const float* __restrict__ Whh,
                                 const float* __restrict__ bih,
                                 const float* __restrict__ bhh,
                                 __bf16* __restrict__ o_tdh,
                                 __bf16* __restrict__ o_hist,
                                 __bf16* __restrict__ o_feat,
                                 __bf16* __restrict__ o_comb,
                                 __bf16* __restrict__ o_ih,
                                 __bf16* __restrict__ o_hh,
                                 float* __restrict__ o_bg) {
  int i0 = blockIdx.x * blockDim.x + threadIdx.x;
  int st = gridDim.x * blockDim.x;
  for (int k = i0; k < H_ * D_; k += st) o_tdh[k] = (__bf16)Wtdh[k];
  for (int k = i0; k < D_ * H_; k += st) o_hist[k] = (__bf16)Whist[k];
  for (int k = i0; k < D_ * D_; k += st) {
    int n = k >> 7, c = k & 127;   // off-diagonal mask folded in
    o_feat[k] = (__bf16)((n == c) ? 0.f : Wfeat[k]);
  }
  for (int k = i0; k < D_ * 2 * D_; k += st) o_comb[k] = (__bf16)Wcomb[k];
  for (int k = i0; k < 4 * H_ * 2 * D_; k += st) o_ih[k] = (__bf16)Wih[k];
  for (int k = i0; k < 4 * H_ * H_; k += st) o_hh[k] = (__bf16)Whh[k];
  for (int k = i0; k < 4 * H_; k += st) o_bg[k] = bih[k] + bhh[k];
}

// ---------------- main scan: one WG per 16 batch rows, loops over T ---------
__global__ __launch_bounds__(256)
void rits_scan_kernel(const float* __restrict__ values,
                      const float* __restrict__ masks,
                      const float* __restrict__ deltas,
                      const float* __restrict__ W_td_x,
                      const float* __restrict__ b_td_x,
                      const float* __restrict__ b_td_h,
                      const float* __restrict__ b_hist,
                      const float* __restrict__ b_feat,
                      const float* __restrict__ b_comb,
                      const __bf16* __restrict__ w_tdh,   // (H,D)   row-major
                      const __bf16* __restrict__ w_hist,  // (D,H)
                      const __bf16* __restrict__ w_feat,  // (D,D) off-diag
                      const __bf16* __restrict__ w_comb,  // (D,2D)
                      const __bf16* __restrict__ w_ih,    // (4H,2D)
                      const __bf16* __restrict__ w_hh,    // (4H,H)
                      const float* __restrict__ b_gates,  // b_ih+b_hh (4H)
                      float* __restrict__ out) {
  // A-panel: 16 rows x 512 K of [ c_c (0..127) | m (128..255) | h_scaled (256..511) ]
  __shared__ __align__(16) __bf16 sh_A5[16 * 512];
  __shared__ __align__(16) __bf16 sh_gx[16 * 128];  // gamma_x bf16
  __shared__ __align__(16) __bf16 sh_d [16 * 128];  // deltas bf16
  __shared__ __align__(16) __bf16 sh_xc[16 * 128];  // x_c bf16
  __shared__ __align__(16) float  sh_x [16 * 128];
  __shared__ __align__(16) float  sh_m [16 * 128];

  const int tid  = threadIdx.x;
  const int wave = tid >> 5;      // 0..7
  const int lane = tid & 31;
  const int hi   = (lane >> 4) & 1;
  const int lr   = lane & 15;
  const int b0   = blockIdx.x * 16;

  // phase-L slot: 8 contiguous elements per thread (16x128 tile)
  const int e0   = tid * 8;
  const int lrow = e0 >> 7;
  const int lcol = e0 & 127;

  // hoisted per-thread constants (small, intentionally kept in registers)
  float diagw_r[8], btx_r[8];
#pragma unroll
  for (int i = 0; i < 8; ++i) {
    int c = lcol + i;
    diagw_r[i] = W_td_x[(size_t)c * D_ + c];
    btx_r[i]   = b_td_x[c];
  }
  const int colw = wave * 16 + lr;          // N column for 8-N-tile GEMMs
  float bth_r[2], bg_r[8];
#pragma unroll
  for (int j = 0; j < 2; ++j) bth_r[j] = b_td_h[(wave + 8 * j) * 16 + lr];
#pragma unroll
  for (int j = 0; j < 8; ++j) bg_r[j] = b_gates[(wave + 8 * j) * 16 + lr];
  const float bhist_r = b_hist[colw];
  const float bfeat_r = b_feat[colw];
  const float bcomb_r = b_comb[colw];

  // opaque zero offset, re-opaqued per phase (anti-LICM, see OPQ above)
  int zoff = 0;

  // register-resident LSTM state: wave w owns h/c columns of N-tiles {w, w+8}
  float h_reg[2][8], c_reg[2][8];
#pragma unroll
  for (int p = 0; p < 2; ++p)
#pragma unroll
    for (int g = 0; g < 8; ++g) { h_reg[p][g] = 0.f; c_reg[p][g] = 0.f; }

  float xh_r[8], al_r[8], m_r[8], x_r[8];
  const v8f vzero = {0.f, 0.f, 0.f, 0.f, 0.f, 0.f, 0.f, 0.f};

#pragma unroll 1
  for (int t = 0; t < T_; ++t) {
    // ---- Phase L: load x/m/d tile, gamma_x, stage bf16 operands ----
    {
      size_t gbase = ((size_t)(b0 + lrow) * T_ + t) * D_ + lcol;
      float xa[8], ma[8], da[8];
      *(float4*)&xa[0] = *(const float4*)(values + gbase);
      *(float4*)&xa[4] = *(const float4*)(values + gbase + 4);
      *(float4*)&ma[0] = *(const float4*)(masks + gbase);
      *(float4*)&ma[4] = *(const float4*)(masks + gbase + 4);
      *(float4*)&da[0] = *(const float4*)(deltas + gbase);
      *(float4*)&da[4] = *(const float4*)(deltas + gbase + 4);
#pragma unroll
      for (int i = 0; i < 8; ++i) {
        int e = e0 + i;
        sh_x[e] = xa[i];
        sh_m[e] = ma[i];
        sh_A5[lrow * 512 + 128 + lcol + i] = (__bf16)ma[i];
        sh_d[e] = (__bf16)da[i];
        float gx = __expf(-fmaxf(da[i] * diagw_r[i] + btx_r[i], 0.f));
        sh_gx[e] = (__bf16)gx;
      }
      if (t + 1 < T_) {
        __builtin_prefetch(values + gbase + D_, 0, 3);
        __builtin_prefetch(masks + gbase + D_, 0, 3);
        __builtin_prefetch(deltas + gbase + D_, 0, 3);
      }
    }
    __syncthreads();

    // ---- Phase 1: gamma_h = exp(-relu(d @ Wtdh^T + b)); h *= gamma_h ----
    OPQ(zoff);
#pragma unroll
    for (int j = 0; j < 2; ++j) {
      int tile = wave + 8 * j;
      v8f acc = vzero;
      const __bf16* arow = sh_d + lr * 128;
      const __bf16* bcol = w_tdh + zoff + (size_t)(tile * 16 + lr) * 128;
#pragma unroll
      for (int k0 = 0; k0 < 128; k0 += 32)
        acc = wmma_bf16(load_a(arow, hi, k0), load_b(bcol, hi, k0), acc);
#pragma unroll
      for (int g = 0; g < 8; ++g) {
        float gh = __expf(-fmaxf(acc[g] + bth_r[j], 0.f));
        float hs = h_reg[j][g] * gh;
        sh_A5[(g + 8 * hi) * 512 + 256 + tile * 16 + lr] = (__bf16)hs;
      }
    }
    __syncthreads();

    // ---- Phase 2: x_h = h@Whist^T+b ; alpha = [gamma_x,m]@Wcomb^T+b ; x_c ----
    {
      OPQ(zoff);
      v8f acc2 = vzero, acc4 = vzero;
      const __bf16* ah  = sh_A5 + lr * 512 + 256;            // h_scaled panel
      const __bf16* bch = w_hist + zoff + (size_t)colw * 256;
#pragma unroll 4
      for (int k0 = 0; k0 < 256; k0 += 32)
        acc2 = wmma_bf16(load_a(ah, hi, k0), load_b(bch, hi, k0), acc2);

      const __bf16* ag  = sh_gx + lr * 128;                  // gamma_x (K 0..127)
      const __bf16* am  = sh_A5 + lr * 512 + 128;            // m (K 128..255)
      const __bf16* bcc = w_comb + zoff + (size_t)colw * 256;
#pragma unroll
      for (int k0 = 0; k0 < 128; k0 += 32)
        acc4 = wmma_bf16(load_a(ag, hi, k0), load_b(bcc, hi, k0), acc4);
#pragma unroll
      for (int k0 = 0; k0 < 128; k0 += 32)
        acc4 = wmma_bf16(load_a(am, hi, k0), load_b(bcc, hi, 128 + k0), acc4);

#pragma unroll
      for (int g = 0; g < 8; ++g) {
        int M = g + 8 * hi;
        xh_r[g] = acc2[g] + bhist_r;
        al_r[g] = acc4[g] + bcomb_r;
        m_r[g]  = sh_m[M * 128 + colw];
        x_r[g]  = sh_x[M * 128 + colw];
        float xc = m_r[g] * x_r[g] + (1.f - m_r[g]) * xh_r[g];
        sh_xc[M * 128 + colw] = (__bf16)xc;
      }
    }
    __syncthreads();

    // ---- Phase 3: z_h = x_c@w_feat^T+b ; blend -> c_c ; emit output ----
    {
      OPQ(zoff);
      v8f acc3 = vzero;
      const __bf16* ax  = sh_xc + lr * 128;
      const __bf16* bcf = w_feat + zoff + (size_t)colw * 128;
#pragma unroll
      for (int k0 = 0; k0 < 128; k0 += 32)
        acc3 = wmma_bf16(load_a(ax, hi, k0), load_b(bcf, hi, k0), acc3);
#pragma unroll
      for (int g = 0; g < 8; ++g) {
        int M = g + 8 * hi;
        float zh = acc3[g] + bfeat_r;
        float ch = al_r[g] * zh + (1.f - al_r[g]) * xh_r[g];
        float cc = m_r[g] * x_r[g] + (1.f - m_r[g]) * ch;
        out[((size_t)(b0 + M) * T_ + t) * D_ + colw] = cc;
        sh_A5[M * 512 + colw] = (__bf16)cc;                  // c_c panel (K 0..127)
      }
    }
    __syncthreads();

    // ---- Phase 4: gates = [c_c,m,h]@[Wih;Whh]^T + b ; LSTM pointwise ----
    {
      OPQ(zoff);
      v8f acc5[8];
#pragma unroll
      for (int j = 0; j < 8; ++j) acc5[j] = vzero;
      const __bf16* arow = sh_A5 + lr * 512;
#pragma unroll 2
      for (int k0 = 0; k0 < 512; k0 += 32) {
        v16bf a = load_a(arow, hi, k0);
        const __bf16* Wk = (k0 < 256) ? (w_ih + zoff) : (w_hh + zoff);
        int kk = k0 & 255;
#pragma unroll
        for (int j = 0; j < 8; ++j) {
          const __bf16* bcol = Wk + (size_t)((wave + 8 * j) * 16 + lr) * 256;
          acc5[j] = wmma_bf16(a, load_b(bcol, hi, kk), acc5[j]);
        }
      }
      // tiles {w,w+8}=i, {w+16,w+24}=f, {w+32,w+40}=g, {w+48,w+56}=o
#pragma unroll
      for (int p = 0; p < 2; ++p) {
#pragma unroll
        for (int g = 0; g < 8; ++g) {
          float is = sigm(acc5[0 + p][g] + bg_r[0 + p]);
          float fs = sigm(acc5[2 + p][g] + bg_r[2 + p]);
          float gt = tanhf(acc5[4 + p][g] + bg_r[4 + p]);
          float os = sigm(acc5[6 + p][g] + bg_r[6 + p]);
          float cn = fs * c_reg[p][g] + is * gt;
          c_reg[p][g] = cn;
          h_reg[p][g] = os * tanhf(cn);
        }
      }
    }
    __syncthreads();
  }
}

extern "C" void kernel_launch(void* const* d_in, const int* in_sizes, int n_in,
                              void* d_out, int out_size, void* d_ws, size_t ws_size,
                              hipStream_t stream) {
  (void)in_sizes; (void)n_in; (void)out_size; (void)ws_size;
  const float* values = (const float*)d_in[0];
  const float* masks  = (const float*)d_in[1];
  const float* deltas = (const float*)d_in[2];
  const float* W_td_h = (const float*)d_in[3];
  const float* b_td_h = (const float*)d_in[4];
  const float* W_td_x = (const float*)d_in[5];
  const float* b_td_x = (const float*)d_in[6];
  const float* W_hist = (const float*)d_in[7];
  const float* b_hist = (const float*)d_in[8];
  const float* W_feat = (const float*)d_in[9];
  const float* b_feat = (const float*)d_in[10];
  const float* W_comb = (const float*)d_in[11];
  const float* b_comb = (const float*)d_in[12];
  const float* W_ih   = (const float*)d_in[13];
  const float* W_hh   = (const float*)d_in[14];
  const float* b_ih   = (const float*)d_in[15];
  const float* b_hh   = (const float*)d_in[16];
  float* out = (float*)d_out;

  // workspace layout (bytes), total ~1.23 MB of bf16 weights + fused bias
  char* ws = (char*)d_ws;
  __bf16* o_tdh  = (__bf16*)(ws + 0);        // 32768 bf16
  __bf16* o_hist = (__bf16*)(ws + 65536);    // 32768
  __bf16* o_feat = (__bf16*)(ws + 131072);   // 16384
  __bf16* o_comb = (__bf16*)(ws + 163840);   // 32768
  __bf16* o_ih   = (__bf16*)(ws + 229376);   // 262144
  __bf16* o_hh   = (__bf16*)(ws + 753664);   // 262144
  float*  o_bg   = (float*) (ws + 1277952);  // 1024 f32

  rits_prep_kernel<<<dim3(256), dim3(256), 0, stream>>>(
      W_td_h, W_hist, W_feat, W_comb, W_ih, W_hh, b_ih, b_hh,
      o_tdh, o_hist, o_feat, o_comb, o_ih, o_hh, o_bg);

  rits_scan_kernel<<<dim3(B_ / 16), dim3(256), 0, stream>>>(
      values, masks, deltas, W_td_x, b_td_x, b_td_h, b_hist, b_feat, b_comb,
      o_tdh, o_hist, o_feat, o_comb, o_ih, o_hh, o_bg, out);
}
